// DeepWalk_38714835206178
// MI455X (gfx1250) — matrix operand order repaired and checked by reading, not compiled, not run on hardware
//
#include <hip/hip_runtime.h>
#include <hip/hip_bf16.h>
#include <stdint.h>

// ---------------------------------------------------------------------------
// MI455X / gfx1250 implementation.
//   Stage 1 (mlp_kernel):  fused 3-layer node MLP via v_wmma_f32_16x16x32_f16
//   Stage 2 (pool_kernel): segment sum/count (f32 atomics) + segment max
//                          (u32-punned atomicMax; valid since y = relu >= 0)
//   Stage 3 (out_kernel):  [C,40]x[40,16] GEMM via WMMA, K padded 40->64
// Weights are prepacked (prep_kernel) into zero-padded, transposed f16 tables
// so each lane's B-fragment is 16 contiguous halves (2x global_load_b128).
// Global->LDS tile fills exploit that a 16-row x 20-col f32 tile is one
// contiguous 320-float span: 5 branchless float2 loads per lane (coalesced
// global_load_b64) + packed f16 ds_store_b32 -- no per-element execz branches.
// ---------------------------------------------------------------------------

typedef _Float16 h16;
typedef __attribute__((ext_vector_type(2)))  _Float16 h16x2;
typedef __attribute__((ext_vector_type(16))) _Float16 v16h;
typedef __attribute__((ext_vector_type(8)))  float    v8f;

union AF { v16h v; h16   e[16]; };
union CF { v8f  v; float e[8];  };

#define WMMA_F16(A, B, C) \
  __builtin_amdgcn_wmma_f32_16x16x32_f16(false, (A), false, (B), (short)0, (C), false, false)

// A operand (16x32 f16, MxK) from an LDS tile laid out [16 rows][64 halves].
// lane m=lane&15; halves 0..15 -> K = h + (hiHalf?16:0) (matches B fragment).
__device__ __forceinline__ v16h load_a_lds(const h16* p, int lane, int kbase) {
  AF f;
  const int m  = lane & 15;
  const int hk = (lane >= 16) ? 16 : 0;
#pragma unroll
  for (int h = 0; h < 16; ++h) {
    const int k = h + hk + kbase;
    f.e[h] = p[m * 64 + k];
  }
  return f.v;
}

// B operand (32x16 f16, KxN) from a transposed global table WT[n][KS]
// (so B[k][n] = WT[n][k]).  lanes 0-15: K=kbase+0..15, lanes 16-31: K=kbase+16..31.
__device__ __forceinline__ v16h load_b_glb(const h16* __restrict__ wt, int KS,
                                           int ncol0, int kbase, int lane) {
  AF f;
  const int n    = lane & 15;
  const int koff = (lane < 16) ? 0 : 16;
  const h16* row = wt + (size_t)(ncol0 + n) * KS + kbase + koff;
#pragma unroll
  for (int h = 0; h < 16; ++h) f.e[h] = row[h];
  return f.v;
}

// Branchless fill of a [16 rows x 20 cols] f32 tile (contiguous span of 320
// floats at src) into LDS halves at column offset cofs, row stride 64 halves.
// nvalid = number of valid floats remaining from src (>= 1).
__device__ __forceinline__ void fill_tile_f32_to_lds(const float* __restrict__ src,
                                                     h16* L, int cofs, int lane,
                                                     long long nvalid) {
  const float2* s2 = (const float2*)src;                  // span is 8B aligned
  int tmax = (int)(nvalid >> 1);                          // valid float2 count
  if (tmax > 160) tmax = 160;
#pragma unroll
  for (int j = 0; j < 5; ++j) {
    const int t  = j * 32 + lane;                         // 0..159
    const int tc = (t < tmax) ? t : (tmax - 1);
    float2 v = s2[tc];
    if (t >= tmax) { v.x = 0.f; v.y = 0.f; }              // cndmask, no branch
    const int r = t / 10;                                 // flat p=2t: row
    const int c = 2 * (t - r * 10);                       //          col (even)
    h16x2 hv = { (h16)v.x, (h16)v.y };
    *(h16x2*)(L + r * 64 + cofs + c) = hv;                // ds_store_b32
  }
}

// ---------------------------------------------------------------------------
// Weight packing:
//  B1T [48][32]: block-diag layer-1:  cols 0..19 <- W_demo (K rows 0..7),
//                cols 20..39 <- W_purch (K rows 8..19), rest zero.
//  B2T [32][64]: W_feat (60x20) zero-padded to K=64, N=32.
//  B3T [16][64]: W_out  (40x16) zero-padded to K=64.
// ---------------------------------------------------------------------------
__global__ void prep_kernel(const float* __restrict__ Wd, const float* __restrict__ bd,
                            const float* __restrict__ Wp, const float* __restrict__ bp,
                            const float* __restrict__ Wf, const float* __restrict__ bf,
                            const float* __restrict__ Wo,
                            h16* __restrict__ B1T, float* __restrict__ bias1,
                            h16* __restrict__ B2T, float* __restrict__ bias2,
                            h16* __restrict__ B3T) {
  const int t = threadIdx.x;
  for (int idx = t; idx < 48 * 32; idx += 256) {
    const int n = idx >> 5, k = idx & 31;
    float v = 0.f;
    if (n < 20 && k < 8) v = Wd[k * 20 + n];
    else if (n >= 20 && n < 40 && k >= 8 && k < 20) v = Wp[(k - 8) * 20 + (n - 20)];
    B1T[idx] = (h16)v;
  }
  for (int n = t; n < 48; n += 256)
    bias1[n] = (n < 20) ? bd[n] : ((n < 40) ? bp[n - 20] : 0.f);
  for (int idx = t; idx < 32 * 64; idx += 256) {
    const int n = idx >> 6, k = idx & 63;
    B2T[idx] = (h16)((n < 20 && k < 60) ? Wf[k * 20 + n] : 0.f);
  }
  for (int n = t; n < 32; n += 256) bias2[n] = (n < 20) ? bf[n] : 0.f;
  for (int idx = t; idx < 16 * 64; idx += 256) {
    const int n = idx >> 6, k = idx & 63;
    B3T[idx] = (h16)((k < 40) ? Wo[k * 16 + n] : 0.f);
  }
}

__global__ void zero_kernel(uint32_t* __restrict__ p, long long n) {
  const long long i = (long long)blockIdx.x * blockDim.x + threadIdx.x;
  if (i < n) p[i] = 0u;
}

// ---------------------------------------------------------------------------
// Stage 1: fused node MLP. One 16-node tile per wave, grid-stride over tiles.
// Per-wave LDS staging tile: [16 rows][64 halves] (2 KB); 8 waves -> 16 KB.
//   layer-1 A (K=32):  cols 0..19 = dataset_x; cols 20..31 hold stale finite
//     relu values -- safe because B1T rows k=20..31 are exactly zero.
//   layer-2 A (K=64):  cols 0..39 = relu(layer1+bias), 40..59 = x,
//     60..63 = 0 (zeroed once; matches W_feat's [demo|purch|x] ordering).
// ---------------------------------------------------------------------------
__global__ void __launch_bounds__(256) mlp_kernel(
    const float* __restrict__ x, const float* __restrict__ dsx,
    const h16* __restrict__ B1T, const float* __restrict__ bias1,
    const h16* __restrict__ B2T, const float* __restrict__ bias2,
    h16* __restrict__ yh, int N, int tiles, int waveTotal) {
  __shared__ h16 lds[8 * 16 * 64];
  const int lane = threadIdx.x & 31;
  const int wv   = threadIdx.x >> 5;
  h16* L = lds + wv * (16 * 64);
  const int waveId = blockIdx.x * 8 + wv;

  // zero the wave-private tile once: uninitialized LDS could hold NaN bit
  // patterns and NaN*0 = NaN would poison the WMMA accumulators.
  {
    uint32_t* p = (uint32_t*)L;
#pragma unroll
    for (int j = 0; j < 16; ++j) p[j * 32 + lane] = 0u;
  }

  // hoisted B fragments + per-lane biases (uniform across tiles)
  v16h b1[3], b2k0[2], b2k1[2];
  float bs1[3], bs2[2];
#pragma unroll
  for (int t = 0; t < 3; ++t) {
    b1[t]  = load_b_glb(B1T, 32, t * 16, 0, lane);
    bs1[t] = bias1[t * 16 + (lane & 15)];
  }
#pragma unroll
  for (int t = 0; t < 2; ++t) {
    b2k0[t] = load_b_glb(B2T, 64, t * 16, 0, lane);
    b2k1[t] = load_b_glb(B2T, 64, t * 16, 32, lane);
    bs2[t]  = bias2[t * 16 + (lane & 15)];
  }

  asm volatile("s_wait_dscnt 0x0" ::: "memory");

  const int rbase = (lane >= 16) ? 8 : 0;
  const int cl    = lane & 15;
  const long long lim = (long long)N * 20;

  for (int tile = waveId; tile < tiles; tile += waveTotal) {
    const int nb = tile * 16;
    const long long base = (long long)nb * 20;

    // dataset_x -> cols 0..19 ; x -> cols 40..59 (both contiguous 320-f32 spans)
    fill_tile_f32_to_lds(dsx + base, L, 0, lane, lim - base);
    fill_tile_f32_to_lds(x + base, L, 40, lane, lim - base);
    asm volatile("s_wait_dscnt 0x0" ::: "memory");

    const v16h a1 = load_a_lds(L, lane, 0);
    asm volatile("s_wait_dscnt 0x0" ::: "memory");

    // layer 1: three 16-col tiles of the 48-wide block-diag weight
    CF d[3];
#pragma unroll
    for (int t = 0; t < 3; ++t) {
      v8f z = {};
      d[t].v = WMMA_F16(a1, b1[t], z);
    }
    // relu(+bias) write-back into layer-2 A cols 0..39 (DS pipe is in-order,
    // so these stores cannot pass the a1 loads above)
#pragma unroll
    for (int t = 0; t < 3; ++t) {
      const int col = t * 16 + cl;
      if (col < 40) {
#pragma unroll
        for (int r = 0; r < 8; ++r) {
          float v = d[t].e[r] + bs1[t];
          v = v > 0.f ? v : 0.f;
          L[(rbase + r) * 64 + col] = (h16)v;
        }
      }
    }
    asm volatile("s_wait_dscnt 0x0" ::: "memory");

    const v16h a2a = load_a_lds(L, lane, 0);
    const v16h a2b = load_a_lds(L, lane, 32);
    asm volatile("s_wait_dscnt 0x0" ::: "memory");

    // layer 2: K=64 (two k-steps), two 16-col tiles covering 20 outputs
#pragma unroll
    for (int t = 0; t < 2; ++t) {
      v8f z = {};
      v8f acc = WMMA_F16(a2a, b2k0[t], z);
      acc     = WMMA_F16(a2b, b2k1[t], acc);
      CF o; o.v = acc;
      const int col = t * 16 + cl;
      if (col < 20) {
#pragma unroll
        for (int r = 0; r < 8; ++r) {
          const int node = nb + rbase + r;
          if (node < N) {
            float v = o.e[r] + bs2[t];
            v = v > 0.f ? v : 0.f;
            yh[(size_t)node * 20 + col] = (h16)v;   // y >= 0 (relu)
          }
        }
      }
    }
  }
}

// ---------------------------------------------------------------------------
// Stage 2: segment pooling. One thread per (extended row, col) element.
// Tables total ~8.2 MB -> L2 resident on MI455X (192 MB).
// ---------------------------------------------------------------------------
__global__ void pool_kernel(const h16* __restrict__ yh,
                            const int* __restrict__ community,
                            const int* __restrict__ mnodes,
                            const int* __restrict__ midx,
                            float* __restrict__ sums,
                            uint32_t* __restrict__ maxb,
                            float* __restrict__ counts,
                            int N, int M) {
  const long long e = (long long)blockIdx.x * blockDim.x + threadIdx.x;
  const long long total = (long long)(N + M) * 20;
  if (e >= total) return;
  const int i = (int)(e / 20);
  const int c = (int)(e - (long long)i * 20);
  int s, node;
  if (i < N) { s = community[i]; node = i; }
  else       { const int j = i - N; node = mnodes[j]; s = midx[j]; }
  const float v = (float)yh[(size_t)node * 20 + c];
  atomicAdd(&sums[(size_t)s * 20 + c], v);
  atomicMax(&maxb[(size_t)s * 20 + c], __float_as_uint(v)); // v>=0 => bit-monotone
  if (c == 0) atomicAdd(&counts[s], 1.0f);
}

// ---------------------------------------------------------------------------
// Stage 3: out = relu([mean|max] @ W_out + b_out), one 16-community tile/wave.
// ---------------------------------------------------------------------------
__global__ void __launch_bounds__(256) out_kernel(
    const float* __restrict__ sums, const uint32_t* __restrict__ maxb,
    const float* __restrict__ counts, const h16* __restrict__ B3T,
    const float* __restrict__ b_out, float* __restrict__ out, int C, int tiles) {
  __shared__ h16 lds[8 * 16 * 64];
  const int lane = threadIdx.x & 31;
  const int wv   = threadIdx.x >> 5;
  h16* L = lds + wv * (16 * 64);
  const int tile = blockIdx.x * 8 + wv;

  {
    uint32_t* p = (uint32_t*)L;
#pragma unroll
    for (int j = 0; j < 16; ++j) p[j * 32 + lane] = 0u;
  }
  if (tile >= tiles) return;  // wave-uniform: remaining waves keep EXEC all-ones

  const int c0 = tile * 16;
  // mean -> pooled k 0..19, max -> pooled k 20..39 (contiguous 320-elem spans)
  {
    const float2* s2 = (const float2*)(sums + (size_t)c0 * 20);
    const uint2*  m2 = (const uint2*)(maxb + (size_t)c0 * 20);
    int tmax = ((C - c0) * 20) >> 1;
    if (tmax > 160) tmax = 160;
#pragma unroll
    for (int j = 0; j < 5; ++j) {
      const int t  = j * 32 + lane;
      const int tc = (t < tmax) ? t : (tmax - 1);
      float2 sv = s2[tc];
      uint2  mv = m2[tc];
      const int r = tc / 10;
      const int c = 2 * (tc - r * 10);
      float cnt = counts[c0 + r];
      cnt = cnt > 1.f ? cnt : 1.f;
      float mx0 = __uint_as_float(mv.x), mx1 = __uint_as_float(mv.y);
      float mn0 = sv.x / cnt, mn1 = sv.y / cnt;
      if (t >= tmax) { mn0 = mn1 = mx0 = mx1 = 0.f; }
      h16x2 hm = { (h16)mn0, (h16)mn1 };
      h16x2 hx = { (h16)mx0, (h16)mx1 };
      *(h16x2*)(L + r * 64 + c)      = hm;
      *(h16x2*)(L + r * 64 + 20 + c) = hx;
    }
  }
  asm volatile("s_wait_dscnt 0x0" ::: "memory");

  const v16h a0 = load_a_lds(L, lane, 0);
  const v16h a1 = load_a_lds(L, lane, 32);
  const v16h b0 = load_b_glb(B3T, 64, 0, 0, lane);
  const v16h b1 = load_b_glb(B3T, 64, 0, 32, lane);
  asm volatile("s_wait_dscnt 0x0" ::: "memory");

  v8f z = {};
  v8f acc = WMMA_F16(a0, b0, z);
  acc     = WMMA_F16(a1, b1, acc);
  CF o; o.v = acc;

  const int cl = lane & 15;
  const int rbase = (lane >= 16) ? 8 : 0;
  const float bb = b_out[cl];
#pragma unroll
  for (int r = 0; r < 8; ++r) {
    const int row = c0 + rbase + r;
    if (row < C) {
      const float v = o.e[r] + bb;
      out[(size_t)row * 16 + cl] = v > 0.f ? v : 0.f;
    }
  }
}

// ---------------------------------------------------------------------------
extern "C" void kernel_launch(void* const* d_in, const int* in_sizes, int n_in,
                              void* d_out, int out_size, void* d_ws, size_t ws_size,
                              hipStream_t stream) {
  const float* x   = (const float*)d_in[0];
  const float* dsx = (const float*)d_in[1];
  const int* community = (const int*)d_in[2];
  const int* mnodes    = (const int*)d_in[3];
  const int* midx      = (const int*)d_in[4];
  const float* Wd = (const float*)d_in[5];  const float* bd = (const float*)d_in[6];
  const float* Wp = (const float*)d_in[7];  const float* bp = (const float*)d_in[8];
  const float* Wf = (const float*)d_in[9];  const float* bf = (const float*)d_in[10];
  const float* Wo = (const float*)d_in[11]; const float* bo = (const float*)d_in[12];
  float* out = (float*)d_out;

  const int N = in_sizes[0] / 20;
  const int M = in_sizes[3];
  const int C = out_size / 16;

  char* ws = (char*)d_ws;
  size_t off = 0;
  auto take = [&](size_t bytes) {
    size_t o = off;
    off = (off + bytes + 255) & ~((size_t)255);
    return o;
  };
  // pooling tables first (contiguous -> single zeroing pass)
  const size_t oSums = take((size_t)C * 20 * 4);
  const size_t oMaxb = take((size_t)C * 20 * 4);
  const size_t oCnt  = take((size_t)C * 4);
  const size_t oB1   = take(48 * 32 * 2);
  const size_t obs1  = take(48 * 4);
  const size_t oB2   = take(32 * 64 * 2);
  const size_t obs2  = take(32 * 4);
  const size_t oB3   = take(16 * 64 * 2);
  const size_t oY    = take((size_t)N * 20 * 2);
  (void)ws_size; (void)n_in;

  prep_kernel<<<1, 256, 0, stream>>>(Wd, bd, Wp, bp, Wf, bf, Wo,
      (h16*)(ws + oB1), (float*)(ws + obs1),
      (h16*)(ws + oB2), (float*)(ws + obs2),
      (h16*)(ws + oB3));

  const long long zn = (long long)(oCnt + (size_t)C * 4 - oSums) / 4;
  zero_kernel<<<(int)((zn + 255) / 256), 256, 0, stream>>>((uint32_t*)(ws + oSums), zn);

  const int tiles = (N + 15) / 16;
  int blocks = 2048;
  if (tiles < blocks * 8) blocks = (tiles + 7) / 8;
  if (blocks < 1) blocks = 1;
  mlp_kernel<<<blocks, 256, 0, stream>>>(
      x, dsx,
      (const h16*)(ws + oB1), (const float*)(ws + obs1),
      (const h16*)(ws + oB2), (const float*)(ws + obs2),
      (h16*)(ws + oY), N, tiles, blocks * 8);

  const long long pe = (long long)(N + M) * 20;
  pool_kernel<<<(int)((pe + 255) / 256), 256, 0, stream>>>(
      (const h16*)(ws + oY), community, mnodes, midx,
      (float*)(ws + oSums), (uint32_t*)(ws + oMaxb), (float*)(ws + oCnt), N, M);

  const int tiles3 = (C + 15) / 16;
  out_kernel<<<(tiles3 + 7) / 8, 256, 0, stream>>>(
      (const float*)(ws + oSums), (const uint32_t*)(ws + oMaxb),
      (const float*)(ws + oCnt), (const h16*)(ws + oB3), bo, out, C, tiles3);
}
